// MultiheadAttention_27255862461086
// MI455X (gfx1250) — matrix operand order repaired and checked by reading, not compiled, not run on hardware
//
#include <hip/hip_runtime.h>
#include <hip/hip_bf16.h>
#include <math.h>

// ---------------------------------------------------------------------------
// MI455X multihead attention, all matmuls via v_wmma_f32_16x16x32_f16 (wave32)
//   pass 1..2: Q/K projections   (f32 in -> f16 out [token][D])
//   pass 3   : V projection      (f32 in -> f16 out TRANSPOSED [B][H][P][N])
//   pass 4   : flash attention   (f16 in -> f16 out),
//              K/V^T tiles via double-buffered Tensor Data Mover DMA
//   pass 5   : output projection (f16 in -> f32 out)
// ---------------------------------------------------------------------------

typedef __attribute__((ext_vector_type(16))) _Float16     v16h;
typedef __attribute__((ext_vector_type(8)))  float        v8f;
typedef __attribute__((ext_vector_type(4)))  unsigned int v4u;
typedef __attribute__((ext_vector_type(8)))  int          v8i;
typedef __attribute__((ext_vector_type(4)))  int          v4i;

#if defined(__gfx1250__) && __has_builtin(__builtin_amdgcn_tensor_load_to_lds)
#define ATTN_HAS_TDM 1
#else
#define ATTN_HAS_TDM 0
#endif

__device__ inline v8f wmma_f16(v16h a, v16h b, v8f c) {
  return __builtin_amdgcn_wmma_f32_16x16x32_f16(false, a, false, b, (short)0, c,
                                                false, false);
}

// --- WMMA fragment loaders (contiguous k order: k = 16*(lane>>4) + s, wave32) -----------
// A 16x32 f16: lane<16 -> row=lane, K={0..15}; lane>=16 -> K={16..31}
__device__ inline v16h load_frag_a(const _Float16* lds, int ld, int rbase,
                                   int kbase, int lane) {
  const int g = (lane >> 4) & 1;
  const _Float16* rowp = lds + (rbase + (lane & 15)) * ld + kbase;
  union { v16h v; unsigned int u[8]; } f;
#pragma unroll
  for (int i = 0; i < 8; ++i) {
    const int k0 = 2 * i + 16 * g;
    f.u[i] = *reinterpret_cast<const unsigned int*>(rowp + k0);
  }
  return f.v;
}

// B 32x16 f16 from [N][K]-major LDS (element (k,n) = lds[n*ld + k]).
// lane: col = lane&15, K = 2i+{0,1} + 16*(lane>>4) -> 16 contiguous halves
__device__ inline v16h load_frag_b_nk(const _Float16* lds, int ld, int kbase,
                                      int nbase, int lane) {
  const int g = (lane >> 4) & 1;
  const _Float16* rowp = lds + (nbase + (lane & 15)) * ld + kbase + 16 * g;
  union { v16h v; unsigned int u[8]; } f;
#pragma unroll
  for (int i = 0; i < 8; ++i)
    f.u[i] = *reinterpret_cast<const unsigned int*>(rowp + 2 * i);
  return f.v;
}

// --- global->LDS tile loaders (convert f32->f16 in flight) ------------------
__device__ inline void load4_to_lds(const float* g, _Float16* l) {
  float4 t = *reinterpret_cast<const float4*>(g);
  union { uint2 u; _Float16 h[4]; } p;
  p.h[0] = (_Float16)t.x; p.h[1] = (_Float16)t.y;
  p.h[2] = (_Float16)t.z; p.h[3] = (_Float16)t.w;
  *reinterpret_cast<uint2*>(l) = p.u;
}
__device__ inline void load4_to_lds(const _Float16* g, _Float16* l) {
  *reinterpret_cast<uint2*>(l) = *reinterpret_cast<const uint2*>(g);
}

// --- Tensor Data Mover: 2D f16 tile (rows x 64 halves) -> padded LDS --------
// LDS rows padded 128B -> 160B  (pad_interval code 4 = 32 DWORDs,
//                                pad_amount  code 7 =  8 DWORDs)
__device__ inline void tdm_load_tile_f16(unsigned lds_off, const _Float16* gptr,
                                         int row_elems, int rows,
                                         long long row_stride_elems) {
#if ATTN_HAS_TDM
  const unsigned long long ga = (unsigned long long)gptr;
  v4u g0;
  g0.x = 1u;                                     // count=1, user descriptor
  g0.y = lds_off;                                // lds_addr (bytes)
  g0.z = (unsigned)(ga & 0xffffffffu);           // global_addr[31:0]
  g0.w = (unsigned)((ga >> 32) & 0x1ffffffu) | (2u << 30);  // [56:32] | type=2
  v8i g1;
  g1[0] = (1 << 16)        // data_size = 1 (2 bytes)
        | (1 << 20)        // pad_enable
        | (4 << 22)        // pad_interval: 32 DWORDs (128B row)
        | (7 << 25);       // pad_amount:    8 DWORDs (32B pad)
  g1[1] = (row_elems & 0xffff) << 16;                             // dim0 lo
  g1[2] = ((row_elems >> 16) & 0xffff) | ((rows & 0xffff) << 16); // dim0 hi|dim1 lo
  g1[3] = ((rows >> 16) & 0xffff) | ((row_elems & 0xffff) << 16); // dim1 hi|tile_dim0
  g1[4] = rows & 0xffff;                                          // tile_dim1, tile_dim2=0
  g1[5] = (int)(row_stride_elems & 0xffffffffll);                 // dim0_stride lo
  g1[6] = (int)((row_stride_elems >> 32) & 0xffffll);             // dim0_stride hi
  g1[7] = 0;
  v4i z4 = {};
#if __clang_major__ >= 23
  v8i z8 = {};
  __builtin_amdgcn_tensor_load_to_lds(g0, g1, z4, z4, z8, 0);
#else
  __builtin_amdgcn_tensor_load_to_lds(g0, g1, z4, z4, 0);
#endif
#else
  (void)lds_off; (void)gptr; (void)row_elems; (void)rows; (void)row_stride_elems;
#endif
}

// ---------------------------------------------------------------------------
// GEMM: Y[M,N] = X[M,K] @ W[N,K]^T + bias
// TRANS=false: Y row-major [M][N]
// TRANS=true : Y = per-head transposed f16 [B][H][PDIM][seq]  (for V)
// block = 256 threads = 8 waves, tile 128x128, K-step 32
// ---------------------------------------------------------------------------
template <typename TIN, typename TOUT, bool TRANS>
__global__ void __launch_bounds__(256)
proj_gemm_kernel(const TIN* __restrict__ X, const float* __restrict__ W,
                 const float* __restrict__ bias, TOUT* __restrict__ Y,
                 int M, int N, int K, int seq) {
  __shared__ _Float16 As[128][40];  // 80B row stride
  __shared__ _Float16 Bs[128][40];

  const int tid  = threadIdx.x;
  const int lane = tid & 31;
  const int wave = tid >> 5;
  const int wr = wave & 3;
  const int wc = wave >> 2;
  const int g  = (lane >> 4) & 1;
  const int cl = lane & 15;

  const int rowbase = blockIdx.y * 128;
  const int colbase = blockIdx.x * 128;

  union Acc { v8f v; float f[8]; };
  Acc acc[2][4];
#pragma unroll
  for (int i = 0; i < 2; ++i)
#pragma unroll
    for (int j = 0; j < 4; ++j)
#pragma unroll
      for (int e = 0; e < 8; ++e) acc[i][j].f[e] = 0.0f;

  const int lr0 = tid >> 3;
  const int lc4 = (tid & 7) * 4;

  for (int k0 = 0; k0 < K; k0 += 32) {
    __syncthreads();
#pragma unroll
    for (int it = 0; it < 4; ++it) {
      const int r = lr0 + it * 32;
      load4_to_lds(X + (size_t)(rowbase + r) * K + k0 + lc4, &As[r][lc4]);
      load4_to_lds(W + (size_t)(colbase + r) * K + k0 + lc4, &Bs[r][lc4]);
    }
    if (k0 + 32 < K)
      __builtin_prefetch(X + (size_t)(rowbase + lr0) * K + k0 + 32 + lc4, 0, 1);
    __syncthreads();

    v16h a0 = load_frag_a(&As[0][0], 40, wr * 32,      0, lane);
    v16h a1 = load_frag_a(&As[0][0], 40, wr * 32 + 16, 0, lane);
#pragma unroll
    for (int ct = 0; ct < 4; ++ct) {
      v16h b = load_frag_b_nk(&Bs[0][0], 40, 0, wc * 64 + ct * 16, lane);
      acc[0][ct].v = wmma_f16(a0, b, acc[0][ct].v);
      acc[1][ct].v = wmma_f16(a1, b, acc[1][ct].v);
    }
  }

#pragma unroll
  for (int rt = 0; rt < 2; ++rt)
#pragma unroll
    for (int ct = 0; ct < 4; ++ct) {
      const int col = colbase + wc * 64 + ct * 16 + cl;
      const float bv = bias[col];
      if (TRANS) {
        // lane owns 8 CONSECUTIVE tokens -> one b128 store, transposed layout
        const int t0   = rowbase + wr * 32 + rt * 16 + 8 * g;  // first token
        const int bidx = t0 / seq;
        const int n0   = t0 - bidx * seq;
        const int head = col >> 6;             // PDIM = 64
        const int p    = col & 63;
        union { uint4 u; _Float16 h[8]; } pk;
#pragma unroll
        for (int r = 0; r < 8; ++r) pk.h[r] = (_Float16)(acc[rt][ct].f[r] + bv);
        _Float16* dst = (_Float16*)Y +
            (((size_t)bidx * 16 + head) * 64 + p) * seq + n0;
        *reinterpret_cast<uint4*>(dst) = pk.u;
      } else {
#pragma unroll
        for (int r = 0; r < 8; ++r) {
          const int row = rowbase + wr * 32 + rt * 16 + r + 8 * g;
          Y[(size_t)row * N + col] = (TOUT)(acc[rt][ct].f[r] + bv);
        }
      }
    }
}

// ---------------------------------------------------------------------------
// Flash attention: one block per (batch, head, 64-row query tile)
// Q,K: [b][token][D] f16;  Vt: [B][H][PDIM][N] f16;  O: [b][token][D] f16
// K/V^T tiles double-buffered: TDM DMA of tile jt+1 overlaps compute of jt.
// ---------------------------------------------------------------------------
__global__ void __launch_bounds__(256)
flash_attn_kernel(const _Float16* __restrict__ Qh, const _Float16* __restrict__ Kh,
                  const _Float16* __restrict__ Vt, _Float16* __restrict__ Oh,
                  int N, int M) {
  constexpr int D = 1024;
  constexpr int PDIM = 64;
  constexpr int H = 16;
  __shared__ _Float16 Qs[64][80];       // 160B row stride (TDM pad-compatible)
  __shared__ _Float16 Ks[2][64][80];    // double-buffered
  __shared__ _Float16 Vs[2][64][80];    // V^T tile: [p][j], double-buffered
  __shared__ _Float16 Ps[64][80];
  __shared__ float    Ss[64][66];
  __shared__ float    row_a[64];

  const int tid  = threadIdx.x;
  const int lane = tid & 31;
  const int wave = tid >> 5;
  const int wr = wave & 3;
  const int wc = wave >> 2;
  const int g  = (lane >> 4) & 1;
  const int cl = lane & 15;

  const int qbase = blockIdx.x * 64;
  const int head  = blockIdx.y;
  const int b     = blockIdx.z;
  const size_t qoff  = ((size_t)b * N + qbase) * D + head * PDIM;
  const size_t kbse  = (size_t)b * M * D + head * PDIM;
  const size_t vtbse = ((size_t)b * H + head) * PDIM * (size_t)M;  // row stride M

  // Q tile -> LDS, pre-scaled by 1/sqrt(PDIM) (exact power of two in f16)
#pragma unroll
  for (int it = 0; it < 2; ++it) {
    const int idx = tid + it * 256;
    const int row = idx >> 3, seg = idx & 7;
    union { uint4 u; _Float16 h[8]; } t;
    t.u = *reinterpret_cast<const uint4*>(Qh + qoff + (size_t)row * D + seg * 8);
#pragma unroll
    for (int e = 0; e < 8; ++e) t.h[e] = t.h[e] * (_Float16)0.125f;
    *reinterpret_cast<uint4*>(&Qs[row][seg * 8]) = t.u;
  }

  float r_m = -1e30f, r_l = 0.0f;
  union Acc { v8f v; float f[8]; };
  Acc o0, o1;
#pragma unroll
  for (int e = 0; e < 8; ++e) { o0.f[e] = 0.0f; o1.f[e] = 0.0f; }

  const int ntiles = M / 64;

#if ATTN_HAS_TDM
  const unsigned ks_off[2] = {(unsigned)(unsigned long long)&Ks[0][0][0],
                              (unsigned)(unsigned long long)&Ks[1][0][0]};
  const unsigned vs_off[2] = {(unsigned)(unsigned long long)&Vs[0][0][0],
                              (unsigned)(unsigned long long)&Vs[1][0][0]};
  // prologue: DMA tile 0 into buffer 0
  if (wave == 0)      tdm_load_tile_f16(ks_off[0], Kh + kbse,  64, 64, D);
  else if (wave == 1) tdm_load_tile_f16(vs_off[0], Vt + vtbse, 64, 64, M);
#endif

  for (int jt = 0; jt < ntiles; ++jt) {
    const int buf = jt & 1;
    __syncthreads();  // all waves done reading the buffer the next DMA targets
#if ATTN_HAS_TDM
    if (jt + 1 < ntiles) {
      if (wave == 0)
        tdm_load_tile_f16(ks_off[buf ^ 1],
                          Kh + kbse + (size_t)(jt + 1) * 64 * D, 64, 64, D);
      else if (wave == 1)
        tdm_load_tile_f16(vs_off[buf ^ 1],
                          Vt + vtbse + (size_t)(jt + 1) * 64, 64, 64, M);
      if (wave < 2) __builtin_amdgcn_s_wait_tensorcnt(1);  // tile jt complete
    } else {
      if (wave < 2) __builtin_amdgcn_s_wait_tensorcnt(0);
    }
#else
#pragma unroll
    for (int it = 0; it < 2; ++it) {
      const int idx = tid + it * 256;
      const int row = idx >> 3, seg = idx & 7;
      *reinterpret_cast<uint4*>(&Ks[buf][row][seg * 8]) =
          *reinterpret_cast<const uint4*>(Kh + kbse + ((size_t)jt * 64 + row) * D + seg * 8);
      *reinterpret_cast<uint4*>(&Vs[buf][row][seg * 8]) =
          *reinterpret_cast<const uint4*>(Vt + vtbse + (size_t)row * M + jt * 64 + seg * 8);
    }
#endif
    __syncthreads();

    const _Float16* ks = &Ks[buf][0][0];
    const _Float16* vs = &Vs[buf][0][0];

    // S = (Q/temp) K^T  (each wave: 16 rows x 32 cols of the 64x64 tile)
    Acc s0, s1;
#pragma unroll
    for (int e = 0; e < 8; ++e) { s0.f[e] = 0.0f; s1.f[e] = 0.0f; }
#pragma unroll
    for (int kk = 0; kk < 64; kk += 32) {
      v16h a  = load_frag_a(&Qs[0][0], 80, wr * 16, kk, lane);
      v16h b0 = load_frag_b_nk(ks, 80, kk, wc * 32,      lane);
      v16h b1 = load_frag_b_nk(ks, 80, kk, wc * 32 + 16, lane);
      s0.v = wmma_f16(a, b0, s0.v);
      s1.v = wmma_f16(a, b1, s1.v);
    }
#pragma unroll
    for (int r = 0; r < 8; ++r) {
      Ss[wr * 16 + r + 8 * g][wc * 32 + cl]      = s0.f[r];
      Ss[wr * 16 + r + 8 * g][wc * 32 + 16 + cl] = s1.f[r];
    }
    __syncthreads();

    // online softmax: thread t<64 owns score row t for the whole loop
    if (tid < 64) {
      float tmax = -1e30f;
      for (int c = 0; c < 64; ++c) tmax = fmaxf(tmax, Ss[tid][c]);
      const float mnew  = fmaxf(r_m, tmax);
      const float alpha = __expf(r_m - mnew);
      float lsum = 0.0f;
      for (int c = 0; c < 64; ++c) {
        const float e = __expf(Ss[tid][c] - mnew);
        Ps[tid][c] = (_Float16)e;
        lsum += e;
      }
      r_l = r_l * alpha + lsum;
      r_m = mnew;
      row_a[tid] = alpha;
    }
    __syncthreads();

    // rescale O accum, then O += P V   (B operand from V^T tile, [n=p][k=j])
#pragma unroll
    for (int r = 0; r < 8; ++r) {
      const float alpha = row_a[wr * 16 + r + 8 * g];
      o0.f[r] *= alpha;
      o1.f[r] *= alpha;
    }
#pragma unroll
    for (int kk = 0; kk < 64; kk += 32) {
      v16h a  = load_frag_a(&Ps[0][0], 80, wr * 16, kk, lane);
      v16h b0 = load_frag_b_nk(vs, 80, kk, wc * 32,      lane);
      v16h b1 = load_frag_b_nk(vs, 80, kk, wc * 32 + 16, lane);
      o0.v = wmma_f16(a, b0, o0.v);
      o1.v = wmma_f16(a, b1, o1.v);
    }
  }

  __syncthreads();
  if (tid < 64) row_a[tid] = 1.0f / r_l;
  __syncthreads();

#pragma unroll
  for (int r = 0; r < 8; ++r) {
    const int row = wr * 16 + r + 8 * g;
    const float inv = row_a[row];
    const size_t base = ((size_t)b * N + qbase + row) * D + head * PDIM;
    Oh[base + wc * 32 + cl]      = (_Float16)(o0.f[r] * inv);
    Oh[base + wc * 32 + 16 + cl] = (_Float16)(o1.f[r] * inv);
  }
}

// ---------------------------------------------------------------------------
extern "C" void kernel_launch(void* const* d_in, const int* in_sizes, int n_in,
                              void* d_out, int out_size, void* d_ws, size_t ws_size,
                              hipStream_t stream) {
  const float* q  = (const float*)d_in[0];
  const float* k  = (const float*)d_in[1];
  const float* v  = (const float*)d_in[2];
  const float* Wq = (const float*)d_in[3];
  const float* bq = (const float*)d_in[4];
  const float* Wk = (const float*)d_in[5];
  const float* bk = (const float*)d_in[6];
  const float* Wv = (const float*)d_in[7];
  const float* bv = (const float*)d_in[8];
  const float* Wo = (const float*)d_in[9];
  const float* bo = (const float*)d_in[10];

  const int D = 1024;
  const int B = 2;
  const int N = in_sizes[0] / (B * D);  // 2048
  const int M = in_sizes[1] / (B * D);  // 2048
  const int H = 16;
  const int tokens = B * N;             // 4096

  _Float16* Qh  = (_Float16*)d_ws;
  _Float16* Kh  = Qh + (size_t)tokens * D;
  _Float16* Vth = Kh + (size_t)tokens * D;   // transposed: [B][H][64][M]
  _Float16* Oh  = Vth + (size_t)B * H * 64 * M;

  dim3 blk(256);
  dim3 gproj(D / 128, tokens / 128);

  proj_gemm_kernel<float, _Float16, false><<<gproj, blk, 0, stream>>>(q, Wq, bq, Qh, tokens, D, D, N);
  proj_gemm_kernel<float, _Float16, false><<<gproj, blk, 0, stream>>>(k, Wk, bk, Kh, tokens, D, D, M);
  proj_gemm_kernel<float, _Float16, true ><<<gproj, blk, 0, stream>>>(v, Wv, bv, Vth, tokens, D, D, M);

  dim3 gattn(N / 64, H, B);
  flash_attn_kernel<<<gattn, blk, 0, stream>>>(Qh, Kh, Vth, Oh, N, M);

  proj_gemm_kernel<_Float16, float, false><<<gproj, blk, 0, stream>>>(Oh, Wo, bo, (float*)d_out, tokens, D, D, N);
}
